// GraphDiveModel_81183471829679
// MI455X (gfx1250) — matrix operand order repaired and checked by reading, not compiled, not run on hardware
//
#include <hip/hip_runtime.h>
#include <hip/hip_bf16.h>

// ---------------------------------------------------------------------------
// CDNA5 / gfx1250, wave32. Matrix ops via v_wmma_f32_16x16x32_f16,
// tile staging via GLOBAL_LOAD_ASYNC_TO_LDS (ASYNCcnt-tracked DMA) with a
// double-buffered software pipeline (copy of tile t+1 overlaps compute of t).
// ---------------------------------------------------------------------------

typedef __attribute__((ext_vector_type(16))) _Float16 v16h;
typedef __attribute__((ext_vector_type(8)))  float    v8f;

#define HEADS 6
#define HID 30
#define FEAT (HEADS * HID)   // 180
#define IN_DIM 1034
#define NEG_SLOPE 0.2f
#define LN_EPS 1e-5f

#if defined(__HIP_DEVICE_COMPILE__) && \
    __has_builtin(__builtin_amdgcn_global_load_async_to_lds_b128) && \
    __has_builtin(__builtin_amdgcn_global_load_async_to_lds_b32) && \
    __has_builtin(__builtin_amdgcn_s_wait_asynccnt)
#define USE_ASYNC_LDS 1
typedef int v4i_vs __attribute__((vector_size(16)));
#define GPTR128(p) ((__attribute__((address_space(1))) v4i_vs*)(p))
#define LPTR128(p) ((__attribute__((address_space(3))) v4i_vs*)(p))
#define GPTR32(p)  ((__attribute__((address_space(1))) int*)(p))
#define LPTR32(p)  ((__attribute__((address_space(3))) int*)(p))
#else
#define USE_ASYNC_LDS 0
#endif

// ---------------- ordered-float encoding for unsigned atomicMax -------------
__device__ __forceinline__ unsigned enc_f(float f) {
    unsigned u = __float_as_uint(f);
    return (u & 0x80000000u) ? ~u : (u | 0x80000000u);
}
__device__ __forceinline__ float dec_f(unsigned e) {
    unsigned u = (e & 0x80000000u) ? (e & 0x7FFFFFFFu) : ~e;
    return __uint_as_float(u);
}

// ---------------------------------------------------------------------------
// Generic tiled GEMM:  C[M,N] = act(A[M,K] @ W[K,N] + bias)
// Block = 256 threads = 8 wave32 waves; block tile 64x64; K step 32.
// LDS holds f32 tiles: Af row-major [row][k], Bt transposed [col][k].
// f32->f16 conversion at fragment build (co-executes with XDL WMMA).
// ---------------------------------------------------------------------------
#define BM 64
#define BN 64
#define BK 32
#define AST 36   // f32 LDS stride; 144B row pitch keeps 16B-aligned chunks
#define NASYNC 10  // async ops per thread per fast-staged tile (2 b128 + 8 b32)

__global__ __launch_bounds__(256) void gemm_wmma_f16(
    const float* __restrict__ A, int lda,
    const float* __restrict__ W, int ldw,
    const float* __restrict__ bias,
    float* __restrict__ C, int ldc,
    int M, int N, int K, int act)
{
    __shared__ float Af0[BM * AST];
    __shared__ float Af1[BM * AST];
    __shared__ float Bt0[BN * AST];
    __shared__ float Bt1[BN * AST];

    const int tid   = threadIdx.x;
    const int lane  = tid & 31;          // wave32
    const int wv    = tid >> 5;          // 0..7
    const int mBase = blockIdx.x * BM;
    const int nBase = blockIdx.y * BN;
    const int rm    = (wv & 3) * 16;     // wave row strip in block
    const int cn0   = (wv >> 2) * 32;    // wave col base (two 16-wide tiles)
    const int lcol  = lane & 15;
    const int akb   = (lane & 16) ? 16 : 0;   // A-frag K base (contiguous, matches B)
    const int bkb   = (lane & 16) ? 16 : 0;   // B-frag K base (contiguous, matches A)

    const bool mFull = (mBase + BM <= M);
    const bool nFull = (nBase + BN <= N);

    v8f acc0 = {};
    v8f acc1 = {};

    // staging coordinates (each thread moves 8 contiguous f32)
    const int arow = tid >> 2;           // 0..63
    const int acb  = (tid & 3) * 8;      // 0,8,16,24
    const int bk   = tid >> 3;           // 0..31
    const int bcb  = (tid & 7) * 8;      // 0..56

    // ---- stage one K-tile of A into AfB ----
    auto stageA = [&](int k0, float* AfB) {
        const int gr = mBase + arow;
        float* dst = &AfB[arow * AST + acb];
        if (mFull && (k0 + BK <= K)) {
            const float* ap = A + (size_t)gr * lda + (k0 + acb);
#if USE_ASYNC_LDS
            __builtin_amdgcn_global_load_async_to_lds_b128(GPTR128(ap),     LPTR128(dst),     0, 0);
            __builtin_amdgcn_global_load_async_to_lds_b128(GPTR128(ap + 4), LPTR128(dst + 4), 0, 0);
#else
#pragma unroll
            for (int j = 0; j < 8; ++j) dst[j] = ap[j];
#endif
        } else {
            const int grc = (gr < M) ? gr : (M - 1);
            const float* ar = A + (size_t)grc * lda;
            float tmp[8];
#pragma unroll
            for (int j = 0; j < 8; ++j) {
                const int kk  = k0 + acb + j;
                const int kkc = (kk < K) ? kk : (K - 1);
                float v = ar[kkc];                        // clamped, unguarded load
                tmp[j] = (gr < M && kk < K) ? v : 0.0f;   // select, no exec dance
            }
#pragma unroll
            for (int j = 0; j < 8; ++j) dst[j] = tmp[j];
        }
    };
    // ---- stage one K-tile of W (transposed) into BtB ----
    auto stageB = [&](int k0, float* BtB) {
        const int gk = k0 + bk;
        if (nFull && (k0 + BK <= K)) {
            const float* wp = W + (size_t)gk * ldw + (nBase + bcb);
#pragma unroll
            for (int j = 0; j < 8; ++j) {
                float* dst = &BtB[(bcb + j) * AST + bk];
#if USE_ASYNC_LDS
                __builtin_amdgcn_global_load_async_to_lds_b32(GPTR32(wp + j), LPTR32(dst), 0, 0);
#else
                *dst = wp[j];
#endif
            }
        } else {
            const int gkc = (gk < K) ? gk : (K - 1);
            const float* wr = W + (size_t)gkc * ldw;
            float tmp[8];
#pragma unroll
            for (int j = 0; j < 8; ++j) {
                const int gc  = nBase + bcb + j;
                const int gcc = (gc < N) ? gc : (N - 1);
                float v = wr[gcc];
                tmp[j] = (gk < K && gc < N) ? v : 0.0f;
            }
#pragma unroll
            for (int j = 0; j < 8; ++j) BtB[(bcb + j) * AST + bk] = tmp[j];
        }
    };
    // ---- fragments (contiguous k = 16*(lane/16) + s in A and B) + 2 WMMA ----
    auto computeTile = [&](const float* AfB, const float* BtB) {
        v16h af;
        const float* ar = &AfB[(rm + lcol) * AST];
#pragma unroll
        for (int i = 0; i < 8; ++i) {
            af[i]     = (_Float16)ar[akb + i];        // K = akb .. akb+7
            af[8 + i] = (_Float16)ar[akb + 8 + i];    // K = akb+8 .. akb+15
        }
        v16h bf0, bf1;
        const float* b0 = &BtB[(cn0 + lcol) * AST + bkb];
        const float* b1 = &BtB[(cn0 + 16 + lcol) * AST + bkb];
#pragma unroll
        for (int i = 0; i < 16; ++i) {
            bf0[i] = (_Float16)b0[i];
            bf1[i] = (_Float16)b1[i];
        }
        acc0 = __builtin_amdgcn_wmma_f32_16x16x32_f16(
            false, af, false, bf0, (short)0, acc0, false, false);
        acc1 = __builtin_amdgcn_wmma_f32_16x16x32_f16(
            false, af, false, bf1, (short)0, acc1, false, false);
    };

    if (mFull && nFull) {
        // ---- double-buffered async pipeline (interior blocks) ----
        const int T = (K + BK - 1) / BK;
        stageA(0, Af0); stageB(0, Bt0);
        for (int t = 0; t < T; ++t) {
            const int nt = t + 1;
            bool nextAsync = false;
            if (nt < T) {
                stageA(nt * BK, (nt & 1) ? Af1 : Af0);   // DMA overlaps compute(t)
                stageB(nt * BK, (nt & 1) ? Bt1 : Bt0);
                nextAsync = (nt * BK + BK <= K);          // tile nt fast-staged?
            }
            (void)nextAsync;
#if USE_ASYNC_LDS
            // async loads complete in order: allowing NASYNC outstanding (tile nt)
            // guarantees tile t's copies have landed.
            if (nextAsync) __builtin_amdgcn_s_wait_asynccnt(NASYNC);
            else           __builtin_amdgcn_s_wait_asynccnt(0);
#endif
            __syncthreads();                      // tile t visible to all waves
            computeTile((t & 1) ? Af1 : Af0, (t & 1) ? Bt1 : Bt0);
            __syncthreads();                      // readers done before buf reuse
        }
    } else {
        // ---- simple single-buffer loop (edge blocks) ----
        for (int k0 = 0; k0 < K; k0 += BK) {
            stageA(k0, Af0); stageB(k0, Bt0);
#if USE_ASYNC_LDS
            __builtin_amdgcn_s_wait_asynccnt(0);
#endif
            __syncthreads();
            computeTile(Af0, Bt0);
            __syncthreads();
        }
    }

    // ---- store: VGPR r -> row r (lanes 0-15) / row r+8 (lanes 16-31) ----
    const int col0  = nBase + cn0 + lcol;
    const int col1  = col0 + 16;
    const int rbase = mBase + rm + ((lane & 16) ? 8 : 0);
#pragma unroll
    for (int r = 0; r < 8; ++r) {
        const int row = rbase + r;
        if (row < M) {
            if (col0 < N) {
                float v = acc0[r] + (bias ? bias[col0] : 0.0f);
                if (act) v = fmaxf(v, 0.0f);
                C[(size_t)row * ldc + col0] = v;
            }
            if (col1 < N) {
                float v = acc1[r] + (bias ? bias[col1] : 0.0f);
                if (act) v = fmaxf(v, 0.0f);
                C[(size_t)row * ldc + col1] = v;
            }
        }
    }
}

// ---------------------------------------------------------------------------
// Year encoder: h[:,200:202] = relu(x[:,1024:1034] @ Wy + by)
// ---------------------------------------------------------------------------
__global__ void year_enc_kernel(const float* __restrict__ x,
                                const float* __restrict__ Wy,
                                const float* __restrict__ by,
                                float* __restrict__ h, int nN)
{
    int n = blockIdx.x * blockDim.x + threadIdx.x;
    if (n >= nN) return;
    const float* xr = x + (size_t)n * IN_DIM + 1024;
#pragma unroll
    for (int j = 0; j < 2; ++j) {
        float t = by[j];
#pragma unroll
        for (int k = 0; k < 10; ++k) t += xr[k] * Wy[k * 2 + j];
        h[(size_t)n * 202 + 200 + j] = fmaxf(t, 0.0f);
    }
}

// ---------------------------------------------------------------------------
// GAT per-node attention coefficients
// ---------------------------------------------------------------------------
__global__ void gat_node_att(const float* __restrict__ hW,
                             const float* __restrict__ att_src,
                             const float* __restrict__ att_dst,
                             float* __restrict__ asrc,
                             float* __restrict__ adst, int nN)
{
    int t = blockIdx.x * blockDim.x + threadIdx.x;
    if (t >= nN * HEADS) return;
    int n = t / HEADS, h = t - n * HEADS;
    const float* hp = hW + (size_t)n * FEAT + h * HID;
    float s = 0.0f, d = 0.0f;
#pragma unroll
    for (int c = 0; c < HID; ++c) {
        s += hp[c] * att_src[h * HID + c];
        d += hp[c] * att_dst[h * HID + c];
    }
    asrc[n * HEADS + h] = s;
    adst[n * HEADS + h] = d;
}

__device__ __forceinline__ void edge_sd(const int* ei, int e, int nE, int& s, int& d) {
    if (e < nE) { s = ei[e]; d = ei[nE + e]; }
    else        { s = d = e - nE; }          // self-loops appended
}

__global__ void gat_edge_max(const int* __restrict__ ei,
                             const float* __restrict__ asrc,
                             const float* __restrict__ adst,
                             unsigned* __restrict__ m, int nE, int nN)
{
    int t = blockIdx.x * blockDim.x + threadIdx.x;
    int tot = (nE + nN) * HEADS;
    if (t >= tot) return;
    int e = t / HEADS, h = t - e * HEADS;
    int s, d; edge_sd(ei, e, nE, s, d);
    float v = asrc[s * HEADS + h] + adst[d * HEADS + h];
    v = (v >= 0.0f) ? v : NEG_SLOPE * v;
    atomicMax(&m[d * HEADS + h], enc_f(v));
}

__global__ void gat_edge_sum(const int* __restrict__ ei,
                             const float* __restrict__ asrc,
                             const float* __restrict__ adst,
                             const unsigned* __restrict__ m,
                             float* __restrict__ z, int nE, int nN)
{
    int t = blockIdx.x * blockDim.x + threadIdx.x;
    int tot = (nE + nN) * HEADS;
    if (t >= tot) return;
    int e = t / HEADS, h = t - e * HEADS;
    int s, d; edge_sd(ei, e, nE, s, d);
    float v = asrc[s * HEADS + h] + adst[d * HEADS + h];
    v = (v >= 0.0f) ? v : NEG_SLOPE * v;
    float w = expf(v - dec_f(m[d * HEADS + h]));
    atomicAdd(&z[d * HEADS + h], w);
}

// one wave32 per edge, lanes stride the 180 channels
__global__ void gat_edge_msg(const int* __restrict__ ei,
                             const float* __restrict__ hW,
                             const float* __restrict__ asrc,
                             const float* __restrict__ adst,
                             const unsigned* __restrict__ m,
                             const float* __restrict__ z,
                             float* __restrict__ out, int nE, int nN)
{
    int gid  = blockIdx.x * blockDim.x + threadIdx.x;
    int wave = gid >> 5;
    int lane = gid & 31;
    if (wave >= nE + nN) return;
    int s, d; edge_sd(ei, wave, nE, s, d);
#pragma unroll
    for (int i = 0; i < 6; ++i) {
        int c = lane + 32 * i;
        if (c < FEAT) {
            int h = c / HID;
            float e = asrc[s * HEADS + h] + adst[d * HEADS + h];
            e = (e >= 0.0f) ? e : NEG_SLOPE * e;
            float w = expf(e - dec_f(m[d * HEADS + h]));
            float alpha = w / (z[d * HEADS + h] + 1e-16f);
            atomicAdd(&out[(size_t)d * FEAT + c],
                      hW[(size_t)s * FEAT + c] * alpha);
        }
    }
}

__global__ void gat_finalize(const float* __restrict__ out,
                             const float* __restrict__ b,
                             float* __restrict__ hnew, int nN)
{
    int t = blockIdx.x * blockDim.x + threadIdx.x;
    if (t >= nN * HID) return;
    int n = t / HID, c = t - n * HID;
    float s = 0.0f;
#pragma unroll
    for (int h = 0; h < HEADS; ++h) s += out[(size_t)n * FEAT + h * HID + c];
    hnew[(size_t)n * HID + c] = fmaxf(s * (1.0f / HEADS) + b[c], 0.0f);
}

// ---------------------------------------------------------------------------
// GCN: degrees, normalized scatter, finalize (+ optional LayerNorm)
// ---------------------------------------------------------------------------
__global__ void deg_kernel(const int* __restrict__ ei,
                           float* __restrict__ deg, int nE, int nN)
{
    int e = blockIdx.x * blockDim.x + threadIdx.x;
    if (e >= nE + nN) return;
    int s, d; edge_sd(ei, e, nE, s, d);
    (void)s;
    atomicAdd(&deg[d], 1.0f);
}

__global__ void gcn_scatter(const int* __restrict__ ei,
                            const float* __restrict__ xw,
                            const float* __restrict__ deg,
                            float* __restrict__ out, int nE, int nN)
{
    int gid  = blockIdx.x * blockDim.x + threadIdx.x;
    int wave = gid >> 5;
    int lane = gid & 31;
    if (wave >= nE + nN || lane >= HID) return;
    int s, d; edge_sd(ei, wave, nE, s, d);
    float norm = rsqrtf(fmaxf(deg[s], 1.0f)) * rsqrtf(fmaxf(deg[d], 1.0f));
    atomicAdd(&out[(size_t)d * HID + lane],
              xw[(size_t)s * HID + lane] * norm);
}

__global__ void gcn_finalize(const float* __restrict__ acc,
                             const float* __restrict__ b,
                             const float* __restrict__ g,     // null -> no LN
                             const float* __restrict__ beta,
                             float* __restrict__ h, int nN)
{
    int n = blockIdx.x * blockDim.x + threadIdx.x;
    if (n >= nN) return;
    float v[HID];
#pragma unroll
    for (int c = 0; c < HID; ++c)
        v[c] = fmaxf(acc[(size_t)n * HID + c] + b[c], 0.0f);
    if (g) {
        float mu = 0.0f;
#pragma unroll
        for (int c = 0; c < HID; ++c) mu += v[c];
        mu *= (1.0f / HID);
        float var = 0.0f;
#pragma unroll
        for (int c = 0; c < HID; ++c) { float dd = v[c] - mu; var += dd * dd; }
        var *= (1.0f / HID);
        float is = rsqrtf(var + LN_EPS);
#pragma unroll
        for (int c = 0; c < HID; ++c) v[c] = (v[c] - mu) * is * g[c] + beta[c];
    }
#pragma unroll
    for (int c = 0; c < HID; ++c) h[(size_t)n * HID + c] = v[c];
}

// ---------------------------------------------------------------------------
// Post MLP (30->30->1) + sigmoid, fused per node
// ---------------------------------------------------------------------------
__global__ void post_kernel(const float* __restrict__ h,
                            const float* __restrict__ W1,
                            const float* __restrict__ b1,
                            const float* __restrict__ W2,
                            const float* __restrict__ b2,
                            float* __restrict__ out, int nN)
{
    int n = blockIdx.x * blockDim.x + threadIdx.x;
    if (n >= nN) return;
    float hv[HID];
#pragma unroll
    for (int k = 0; k < HID; ++k) hv[k] = h[(size_t)n * HID + k];
    float o = b2[0];
    for (int j = 0; j < HID; ++j) {
        float t = b1[j];
#pragma unroll
        for (int k = 0; k < HID; ++k) t += hv[k] * W1[k * HID + j];
        o += t * W2[j];
    }
    out[n] = 1.0f / (1.0f + expf(-o));
}

// ---------------------------------------------------------------------------
// Host-side orchestration
// ---------------------------------------------------------------------------
static inline void launch_gemm(const float* A, int lda, const float* W, int ldw,
                               const float* bias, float* C, int ldc,
                               int M, int N, int K, int act, hipStream_t s)
{
    dim3 grid((M + BM - 1) / BM, (N + BN - 1) / BN);
    gemm_wmma_f16<<<grid, 256, 0, s>>>(A, lda, W, ldw, bias, C, ldc, M, N, K, act);
}

extern "C" void kernel_launch(void* const* d_in, const int* in_sizes, int n_in,
                              void* d_out, int out_size, void* d_ws, size_t ws_size,
                              hipStream_t stream)
{
    const int nN = in_sizes[0] / IN_DIM;     // 50000
    const int nE = in_sizes[1] / 2;          // 500000
    const int E  = nE + nN;                  // edges + self-loops

    // inputs in setup_inputs() flatten order
    const float* x   = (const float*)d_in[0];
    const int*   ei  = (const int*)  d_in[1];
    const float* tW0 = (const float*)d_in[2];
    const float* tb0 = (const float*)d_in[3];
    const float* tW1 = (const float*)d_in[4];
    const float* tb1 = (const float*)d_in[5];
    const float* aW0 = (const float*)d_in[6];
    const float* ab0 = (const float*)d_in[7];
    const float* aW1 = (const float*)d_in[8];
    const float* ab1 = (const float*)d_in[9];
    const float* yW  = (const float*)d_in[10];
    const float* yb  = (const float*)d_in[11];
    const float* gatW[3]   = {(const float*)d_in[12], (const float*)d_in[16], (const float*)d_in[20]};
    const float* gatAS[3]  = {(const float*)d_in[13], (const float*)d_in[17], (const float*)d_in[21]};
    const float* gatAD[3]  = {(const float*)d_in[14], (const float*)d_in[18], (const float*)d_in[22]};
    const float* gatB[3]   = {(const float*)d_in[15], (const float*)d_in[19], (const float*)d_in[23]};
    const float* gcnW[2]   = {(const float*)d_in[24], (const float*)d_in[26]};
    const float* gcnB[2]   = {(const float*)d_in[25], (const float*)d_in[27]};
    const float* lnG  = (const float*)d_in[28];
    const float* lnB  = (const float*)d_in[29];
    const float* pW1  = (const float*)d_in[30];
    const float* pb1  = (const float*)d_in[31];
    const float* pW2  = (const float*)d_in[32];
    const float* pb2  = (const float*)d_in[33];
    float* out = (float*)d_out;

    // workspace carve-out
    char* ws = (char*)d_ws;
    size_t off = 0;
    auto carve = [&](size_t bytes) { char* p = ws + off; off = (off + bytes + 255) & ~(size_t)255; return p; };
    float*    big0 = (float*)   carve((size_t)nN * 256 * 4);     // t1/a1 then GAT/GCN accumulators
    float*    big1 = (float*)   carve((size_t)nN * 202 * 4);     // node features h
    float*    big2 = (float*)   carve((size_t)nN * FEAT * 4);    // hW / xw
    float*    asrc = (float*)   carve((size_t)nN * HEADS * 4);
    float*    adst = (float*)   carve((size_t)nN * HEADS * 4);
    unsigned* mEnc = (unsigned*)carve((size_t)nN * HEADS * 4);
    float*    zSum = (float*)   carve((size_t)nN * HEADS * 4);
    float*    deg  = (float*)   carve((size_t)nN * 4);
    (void)ws_size; (void)n_in; (void)out_size;

    const int TPB = 256;
    auto blocks = [&](long long work) { return (int)((work + TPB - 1) / TPB); };

    // ---- feature encoders (WMMA GEMMs) ----
    launch_gemm(x,        IN_DIM, tW0, 256, tb0, big0,       256, nN, 256, 768, 1, stream); // t1
    launch_gemm(big0,     256,    tW1, 100, tb1, big1,       202, nN, 100, 256, 1, stream); // t -> h[:, :100]
    launch_gemm(x + 768,  IN_DIM, aW0, 256, ab0, big0,       256, nN, 256, 256, 1, stream); // a1 (reuse big0)
    launch_gemm(big0,     256,    aW1, 100, ab1, big1 + 100, 202, nN, 100, 256, 1, stream); // a -> h[:,100:200]
    year_enc_kernel<<<blocks(nN), TPB, 0, stream>>>(x, yW, yb, big1, nN);                   // h[:,200:202]

    // ---- degrees (shared by both GCN layers) ----
    (void)hipMemsetAsync(deg, 0, (size_t)nN * 4, stream);
    deg_kernel<<<blocks(E), TPB, 0, stream>>>(ei, deg, nE, nN);

    // ---- 3 GAT layers ----
    for (int l = 0; l < 3; ++l) {
        const int K   = (l == 0) ? 202 : HID;
        const int lda = (l == 0) ? 202 : HID;
        launch_gemm(big1, lda, gatW[l], FEAT, nullptr, big2, FEAT, nN, FEAT, K, 0, stream); // hW
        gat_node_att<<<blocks((long long)nN * HEADS), TPB, 0, stream>>>(
            big2, gatAS[l], gatAD[l], asrc, adst, nN);
        (void)hipMemsetAsync(mEnc, 0, (size_t)nN * HEADS * 4, stream); // enc(0) < enc(any update via self-loop)
        (void)hipMemsetAsync(zSum, 0, (size_t)nN * HEADS * 4, stream);
        (void)hipMemsetAsync(big0, 0, (size_t)nN * FEAT * 4, stream);
        gat_edge_max<<<blocks((long long)E * HEADS), TPB, 0, stream>>>(ei, asrc, adst, mEnc, nE, nN);
        gat_edge_sum<<<blocks((long long)E * HEADS), TPB, 0, stream>>>(ei, asrc, adst, mEnc, zSum, nE, nN);
        gat_edge_msg<<<blocks((long long)E * 32), TPB, 0, stream>>>(
            ei, big2, asrc, adst, mEnc, zSum, big0, nE, nN);
        gat_finalize<<<blocks((long long)nN * HID), TPB, 0, stream>>>(big0, gatB[l], big1, nN);
    }

    // ---- 2 GCN layers (LayerNorm after the first) ----
    for (int i = 0; i < 2; ++i) {
        launch_gemm(big1, HID, gcnW[i], HID, nullptr, big2, HID, nN, HID, HID, 0, stream);  // xw
        (void)hipMemsetAsync(big0, 0, (size_t)nN * HID * 4, stream);
        gcn_scatter<<<blocks((long long)E * 32), TPB, 0, stream>>>(ei, big2, deg, big0, nE, nN);
        gcn_finalize<<<blocks(nN), TPB, 0, stream>>>(
            big0, gcnB[i], (i == 0) ? lnG : nullptr, (i == 0) ? lnB : nullptr, big1, nN);
    }

    // ---- post MLP + sigmoid ----
    post_kernel<<<blocks(nN), TPB, 0, stream>>>(big1, pW1, pb1, pW2, pb2, out, nN);
}
